// MHA_37048387895984
// MI455X (gfx1250) — matrix operand order repaired and checked by reading, not compiled, not run on hardware
//
#include <hip/hip_runtime.h>
#include <hip/hip_bf16.h>

// ---------------------------------------------------------------------------
// Linear attention (elu+1 feature map, causal) for gfx1250 / MI455X.
//   B=2, S=2048, D=1024, H=16, HD=64
// Kernel 1: fused projection GEMM + phi + f32->f16 (double-buffered K loop).
// Kernel 2: chunked causal linear attention, chunk=64, all matmuls via
//           v_wmma_f32_16x16x32_f16 (wave32 WMMA), register-resident state.
// ---------------------------------------------------------------------------

typedef _Float16 half_t;
typedef __attribute__((ext_vector_type(16))) _Float16 v16h;
typedef __attribute__((ext_vector_type(8)))  _Float16 v8h;
typedef __attribute__((ext_vector_type(4)))  _Float16 v4h;
typedef __attribute__((ext_vector_type(4)))  float    v4f;
typedef __attribute__((ext_vector_type(8)))  float    v8f;

#define WMMA_F16(a, b, c) \
  __builtin_amdgcn_wmma_f32_16x16x32_f16(false, (a), false, (b), (short)0, (c), false, false)

// A fragment (16x32, f16), wave32 layout per CDNA5 ISA 7.12.2:
//   lane L holds row (row_base + (L&15));
//   L<16 : halfs = K[k..k+7]   ++ K[k+16..k+23]
//   L>=16: halfs = K[k+8..k+15] ++ K[k+24..k+31]
__device__ __forceinline__ v16h load_frag_a(const half_t* base, int stride,
                                            int row_base, int k_base, int lane) {
  int r  = row_base + (lane & 15);
  int kh = k_base + ((lane & 16) ? 8 : 0);
  const half_t* p = base + r * stride + kh;
  v8h lo = *(const v8h*)(p);
  v8h hi = *(const v8h*)(p + 16);
  v16h a;
#pragma unroll
  for (int i = 0; i < 8; ++i) { a[i] = lo[i]; a[i + 8] = hi[i]; }
  return a;
}

// B fragment (32x16, f16): lane L holds column (n_base + (L&15)),
// K halfs k_base + (L>=16 ? 8 : 0) + [0..7] ++ [16..23] (same as A).
// LDS buffer must be n-major: buf[n * stride + k].
__device__ __forceinline__ v16h load_frag_b(const half_t* base, int stride,
                                            int n_base, int k_base, int lane) {
  int n = n_base + (lane & 15);
  int k = k_base + ((lane & 16) ? 8 : 0);
  const half_t* p = base + n * stride + k;
  v8h lo = *(const v8h*)(p);
  v8h hi = *(const v8h*)(p + 16);
  v16h b;
#pragma unroll
  for (int i = 0; i < 8; ++i) { b[i] = lo[i]; b[i + 8] = hi[i]; }
  return b;
}

// ---------------------------------------------------------------------------
// Kernel 1: OUT[m][n] = sum_k X[m][k] * W[k][n], m in [0,4096), n,k in [0,1024)
// phi(x)=elu(x)+1 (compile-time flag), stored f16 head-major [B*H][S][64].
// grid = (64, 16), block = 128 (4 waves), 64x64 tile, double-buffered K=32 steps.
// ---------------------------------------------------------------------------
template <bool APPLY_PHI>
__global__ __launch_bounds__(128)
void linattn_proj_phi(const float* __restrict__ X, const float* __restrict__ W,
                      half_t* __restrict__ outh) {
  constexpr int LDP = 40;                             // padded stride (halfs)
  __shared__ __align__(16) half_t sa [2][64 * LDP];   // X tile, row-major
  __shared__ __align__(16) half_t sbt[2][64 * LDP];   // W tile, n-major

  const int m0 = blockIdx.x * 64;
  const int n0 = blockIdx.y * 64;
  const int tid = threadIdx.x;
  const int lane = tid & 31, wave = tid >> 5;

  v8f zero = {};
  v8f acc[4];
#pragma unroll
  for (int t = 0; t < 4; ++t) acc[t] = zero;

  v4f xv[4], wv[4];                                   // staged float4s

  auto load_tile = [&](int kk) {
#pragma unroll
    for (int e = 0; e < 4; ++e) {
      int lin = tid + 128 * e;                        // 0..511 float4 groups
      int r = lin >> 3, c4 = (lin & 7) << 2;          // X: 64 rows x 8 groups
      xv[e] = *(const v4f*)(X + (size_t)(m0 + r) * 1024 + kk + c4);
      int kl = lin >> 4, n4 = (lin & 15) << 2;        // W: 32 k x 16 groups
      wv[e] = *(const v4f*)(W + (size_t)(kk + kl) * 1024 + n0 + n4);
    }
  };
  auto store_tile = [&](int buf) {
#pragma unroll
    for (int e = 0; e < 4; ++e) {
      int lin = tid + 128 * e;
      int r = lin >> 3, c4 = (lin & 7) << 2;
      v4h hx;
#pragma unroll
      for (int i = 0; i < 4; ++i) hx[i] = (half_t)xv[e][i];
      *(v4h*)&sa[buf][r * LDP + c4] = hx;             // packed 8B store
      int kl = lin >> 4, n4 = (lin & 15) << 2;
#pragma unroll
      for (int i = 0; i < 4; ++i)
        sbt[buf][(n4 + i) * LDP + kl] = (half_t)wv[e][i];
    }
  };

  load_tile(0);
  store_tile(0);
  __syncthreads();

  for (int t32 = 0; t32 < 32; ++t32) {
    const int buf = t32 & 1;
    if (t32 + 1 < 32) {
      load_tile((t32 + 1) * 32);                      // overlap HBM with WMMA
      __builtin_prefetch(X + (size_t)(m0 + (tid >> 1)) * 1024 + (t32 + 1) * 32, 0, 3);
    }
    v16h a = load_frag_a(sa[buf], LDP, wave * 16, 0, lane);
#pragma unroll
    for (int t = 0; t < 4; ++t)
      acc[t] = WMMA_F16(a, load_frag_b(sbt[buf], LDP, t * 16, 0, lane), acc[t]);
    if (t32 + 1 < 32) store_tile(buf ^ 1);
    __syncthreads();
  }

  // Epilogue: phi + f16 convert + scatter to [B*H][S][64]
#pragma unroll
  for (int t = 0; t < 4; ++t) {
#pragma unroll
    for (int r = 0; r < 8; ++r) {
      int M  = wave * 16 + r + ((lane & 16) ? 8 : 0);
      int gm = m0 + M;                       // row over B*S
      int bb = gm >> 11, s = gm & 2047;
      int N  = n0 + t * 16 + (lane & 15);    // col over H*HD
      int hh = N >> 6, hd = N & 63;
      float v = acc[t][r];
      if (APPLY_PHI) {
        float ev = __expf(fminf(v, 0.0f));   // branchless elu(x)+1
        v = (v > 0.0f) ? (v + 1.0f) : ev;
      }
      outh[((size_t)(bb * 16 + hh) * 2048 + s) * 64 + hd] = (half_t)v;
    }
  }
}

// ---------------------------------------------------------------------------
// Kernel 2: chunked causal linear attention.  One block per (b,h), block=128
// (4 waves; wave w owns rows/features [16w,16w+16)).  32 chunks of 64 rows.
//   P      = tril(phiQ @ phiK^T)                (WMMA, masked in registers)
//   out    = phiQ @ S_prev + P @ V              (WMMA)
//   S     += phiK^T @ V                         (WMMA, reg-resident state)
//   qz[r]  = phiQ[r].z_prev + rowsum(P)[r] + eps
// ---------------------------------------------------------------------------
__global__ __launch_bounds__(128)
void linattn_scan(const half_t* __restrict__ phiq, const half_t* __restrict__ phik,
                  const half_t* __restrict__ vh, float* __restrict__ out) {
  constexpr int LDA = 72;                         // padded stride (halfs)
  __shared__ __align__(16) half_t sq [64 * LDA];  // phiQ chunk, row-major [s][f]
  __shared__ __align__(16) half_t sk [64 * LDA];  // phiK chunk, row-major [s][f]
  __shared__ __align__(16) half_t skt[64 * LDA];  // phiK^T        [f][s]
  __shared__ __align__(16) half_t svt[64 * LDA];  // V^T           [d][s]
  __shared__ __align__(16) half_t sp [64 * LDA];  // masked scores [s][s']
  __shared__ __align__(16) half_t ssh[64 * LDA];  // S_prev n-major [d][f]
  __shared__ float zs[64];                        // running z (cumsum phiK)
  __shared__ float qz[64];                        // per-row normalizer

  const int bh = blockIdx.x;
  const int b = bh >> 4, h = bh & 15;
  const int tid = threadIdx.x, lane = tid & 31, wave = tid >> 5;
  const size_t base_bh = (size_t)bh * 2048 * 64;

  v8f zero = {};
  v8f sacc[4];                                    // state strip S[16w..16w+16)[*]
#pragma unroll
  for (int t = 0; t < 4; ++t) sacc[t] = zero;
  if (tid < 64) zs[tid] = 0.0f;
  __syncthreads();

  for (int chunk = 0; chunk < 32; ++chunk) {
    const size_t cb = base_bh + (size_t)chunk * 64 * 64;

    // ---- stage chunk into LDS as 16B vectors (plus transposes) ----
#pragma unroll
    for (int e = 0; e < 4; ++e) {
      int g = tid + 128 * e;                      // 0..511 8-half groups
      int r = g >> 3, c8 = (g & 7) << 3;
      v8h qv = *(const v8h*)(phiq + cb + (size_t)g * 8);
      v8h kv = *(const v8h*)(phik + cb + (size_t)g * 8);
      v8h vv = *(const v8h*)(vh   + cb + (size_t)g * 8);
      *(v8h*)&sq[r * LDA + c8] = qv;
      *(v8h*)&sk[r * LDA + c8] = kv;
#pragma unroll
      for (int i = 0; i < 8; ++i) {
        skt[(c8 + i) * LDA + r] = kv[i];
        svt[(c8 + i) * LDA + r] = vv[i];
      }
    }
    if (chunk + 1 < 32) {                         // gfx1250 global_prefetch_b8
      const size_t nb = cb + 4096;
      __builtin_prefetch(phiq + nb + (size_t)tid * 32, 0, 3);
      __builtin_prefetch(phik + nb + (size_t)tid * 32, 0, 3);
      __builtin_prefetch(vh   + nb + (size_t)tid * 32, 0, 3);
    }
    __syncthreads();

    // ---- qz[r] = phiQ[r] . z_prev + eps (rowsum of P added below) ----
    if (tid < 64) {
      float a = 0.0f;
      for (int f = 0; f < 64; ++f) a += (float)sq[tid * LDA + f] * zs[f];
      qz[tid] = a + 1e-6f;
    }

    // ---- dump S_prev to LDS in B-matrix layout ssh[d][f] ----
#pragma unroll
    for (int t = 0; t < 4; ++t)
#pragma unroll
      for (int r = 0; r < 8; ++r) {
        int f = wave * 16 + r + ((lane & 16) ? 8 : 0);
        int d = t * 16 + (lane & 15);
        ssh[d * LDA + f] = (half_t)sacc[t][r];
      }

    // ---- P = phiQ @ phiK^T  (B = sk row-major == (phiK^T) n-major) ----
    v8f pacc[4];
#pragma unroll
    for (int t = 0; t < 4; ++t) pacc[t] = zero;
#pragma unroll
    for (int ks = 0; ks < 2; ++ks) {
      v16h a = load_frag_a(sq, LDA, wave * 16, ks * 32, lane);
#pragma unroll
      for (int t = 0; t < 4; ++t)
        pacc[t] = WMMA_F16(a, load_frag_b(sk, LDA, t * 16, ks * 32, lane), pacc[t]);
    }
    __syncthreads();   // qz initialized, ssh visible

    // ---- causal mask + rowsum into qz + spill P to LDS ----
#pragma unroll
    for (int t = 0; t < 4; ++t)
#pragma unroll
      for (int r = 0; r < 8; ++r) {
        int M = wave * 16 + r + ((lane & 16) ? 8 : 0);
        int N = t * 16 + (lane & 15);
        float v = pacc[t][r];
        if (N > M) v = 0.0f;                     // strictly-upper masked out
        atomicAdd(&qz[M], v);                    // ds_add_f32
        sp[M * LDA + N] = (half_t)v;
      }
    __syncthreads();   // sp, qz final

    // ---- out = phiQ @ S_prev + P @ V ----
    v8f oacc[4];
#pragma unroll
    for (int t = 0; t < 4; ++t) oacc[t] = zero;
#pragma unroll
    for (int ks = 0; ks < 2; ++ks) {
      v16h a = load_frag_a(sq, LDA, wave * 16, ks * 32, lane);
#pragma unroll
      for (int t = 0; t < 4; ++t)
        oacc[t] = WMMA_F16(a, load_frag_b(ssh, LDA, t * 16, ks * 32, lane), oacc[t]);
    }
#pragma unroll
    for (int ks = 0; ks < 2; ++ks) {
      v16h a = load_frag_a(sp, LDA, wave * 16, ks * 32, lane);
#pragma unroll
      for (int t = 0; t < 4; ++t)
        oacc[t] = WMMA_F16(a, load_frag_b(svt, LDA, t * 16, ks * 32, lane), oacc[t]);
    }

    // ---- state update S += phiK^T @ V ----
#pragma unroll
    for (int ks = 0; ks < 2; ++ks) {
      v16h a = load_frag_a(skt, LDA, wave * 16, ks * 32, lane);
#pragma unroll
      for (int t = 0; t < 4; ++t)
        sacc[t] = WMMA_F16(a, load_frag_b(svt, LDA, t * 16, ks * 32, lane), sacc[t]);
    }

    // ---- z += colsum(phiK chunk) ----
    if (tid < 64) {
      float s = 0.0f;
      for (int r = 0; r < 64; ++r) s += (float)sk[r * LDA + tid];
      zs[tid] += s;
    }

    // ---- normalize + store: out[b][s][h*64+d] ----
#pragma unroll
    for (int t = 0; t < 4; ++t)
#pragma unroll
      for (int r = 0; r < 8; ++r) {
        int M = wave * 16 + r + ((lane & 16) ? 8 : 0);
        int N = t * 16 + (lane & 15);
        int sg = chunk * 64 + M;
        float val = oacc[t][r] / qz[M];
        out[((size_t)b * 2048 + sg) * 1024 + h * 64 + N] = val;
      }
    __syncthreads();   // protect LDS + zs for next chunk
  }
}

// ---------------------------------------------------------------------------
extern "C" void kernel_launch(void* const* d_in, const int* in_sizes, int n_in,
                              void* d_out, int out_size, void* d_ws, size_t ws_size,
                              hipStream_t stream) {
  (void)in_sizes; (void)n_in; (void)out_size; (void)ws_size;
  const float* q  = (const float*)d_in[0];
  const float* k  = (const float*)d_in[1];
  const float* v  = (const float*)d_in[2];
  const float* wq = (const float*)d_in[3];
  const float* wk = (const float*)d_in[4];
  const float* wv = (const float*)d_in[5];
  float* out = (float*)d_out;

  const size_t per = (size_t)32 * 2048 * 64;     // B*H * S * HD elements
  half_t* phiq = (half_t*)d_ws;
  half_t* phik = phiq + per;
  half_t* vh   = phik + per;

  dim3 pgrid(64, 16), pblk(128);
  linattn_proj_phi<true ><<<pgrid, pblk, 0, stream>>>(q, wq, phiq);
  linattn_proj_phi<true ><<<pgrid, pblk, 0, stream>>>(k, wk, phik);
  linattn_proj_phi<false><<<pgrid, pblk, 0, stream>>>(v, wv, vh);

  linattn_scan<<<dim3(32), dim3(128), 0, stream>>>(phiq, phik, vh, out);
}
